// ParallelLlamaRingAttention_87703232185039
// MI455X (gfx1250) — matrix operand order repaired and checked by reading, not compiled, not run on hardware
//
#include <hip/hip_runtime.h>

// ---------------------------------------------------------------------------
// Types for CDNA5 WMMA (wave32, 16x16x32 bf16 -> f32 accumulate)
// ---------------------------------------------------------------------------
typedef __bf16 v16bf __attribute__((ext_vector_type(16)));
typedef float  v8f   __attribute__((ext_vector_type(8)));
typedef unsigned int u32;
typedef u32 uint4v __attribute__((ext_vector_type(4)));

union FragAB {           // 32 bytes = 8 VGPRs: one WMMA A or B operand
    uint4v u[2];
    v16bf  v;
};

// hardware f32 -> bf16 (RNE) via native conversion; bit pattern as ushort
__device__ __forceinline__ unsigned short to_bf16(float f) {
    union { __bf16 h; unsigned short s; } u;
    u.h = (__bf16)f;
    return u.s;
}
__device__ __forceinline__ float bf16_to_f32(unsigned short h) {
    union { u32 u; float f; } v; v.u = ((u32)h) << 16; return v.f;
}
__device__ __forceinline__ v8f zero_v8() {
    v8f z;
    #pragma unroll
    for (int i = 0; i < 8; ++i) z[i] = 0.0f;
    return z;
}
__device__ __forceinline__ v8f wmma_bf16(const FragAB& a, const FragAB& b, v8f c) {
    return __builtin_amdgcn_wmma_f32_16x16x32_bf16(
        /*neg_a=*/false, a.v, /*neg_b=*/false, b.v,
        /*c_mod=*/(short)0, c, /*reuse_a=*/false, /*reuse_b=*/false);
}
__device__ __forceinline__ float red_max16(float x) {
    #pragma unroll
    for (int off = 1; off < 16; off <<= 1) x = fmaxf(x, __shfl_xor(x, off, 32));
    return x;
}
__device__ __forceinline__ float red_sum16(float x) {
    #pragma unroll
    for (int off = 1; off < 16; off <<= 1) x += __shfl_xor(x, off, 32);
    return x;
}

// ---------------------------------------------------------------------------
// Problem constants
// ---------------------------------------------------------------------------
#define L_SEQ  2048
#define BATCH  2
#define HIDDIM 4096
#define NH     32
#define NKV    8
#define HD     128
#define QKVN   6144          // (NH + 2*NKV) * HD
#define ROWS   4096          // L_SEQ * BATCH
#define ATTN_SCALE 0.08838834764831845f   // 1/sqrt(128)
#define NEG_BIG (-1e30f)

// ---------------------------------------------------------------------------
// Double-buffered tiled bf16 WMMA GEMM:  C[M,N] = A[M,K] * B[K,N]
//   A: fp32 (converted in-flight) or bf16, row-major
//   B: fp32 [K][N] row-major, converted + transposed into LDS
//   Out: bf16 [M][N]   or fp32 with optional (b*L+l)->(l*B+b) row remap
// Block: 256 threads (8 waves), 128x128 tile, BK = 32. Wave tile: 32x64.
// Schedule per K-step: issue next-tile global loads -> 8 WMMA on current
// LDS buffer -> convert/store next tile into the other buffer -> barrier.
// ---------------------------------------------------------------------------
template <bool A_IS_BF16, bool OUT_BF16, bool REMAP_BL>
__global__ __launch_bounds__(256)
void gemm_bf16_wmma(const void* __restrict__ Aptr,
                    const float* __restrict__ B,
                    void* __restrict__ Cptr,
                    int M, int N, int K) {
    constexpr int BM = 128, BN = 128, BK = 32;
    __shared__ unsigned short As[2][BM][BK];   // ping-pong, 16 KB
    __shared__ unsigned short Bs[2][BN][BK];   // ping-pong, 16 KB ([n][k])

    const int tid  = threadIdx.x;
    const int wave = tid >> 5;
    const int lane = tid & 31;
    const int half = lane >> 4;
    const int l16  = lane & 15;

    const int m0 = blockIdx.y * BM;
    const int n0 = blockIdx.x * BN;
    const int wm = (wave & 3) * 32;    // wave M offset in tile
    const int wn = (wave >> 2) * 64;   // wave N offset in tile

    // staging indices
    const int ar = tid >> 1, acoff = (tid & 1) * 16;     // A: 2 thr/row, 16 elems
    const int bk = tid >> 3, bnoff = (tid & 7) * 16;     // B: 8 thr/row, 16 elems

    v8f acc[2][4];
    #pragma unroll
    for (int i = 0; i < 2; ++i)
        #pragma unroll
        for (int j = 0; j < 4; ++j) acc[i][j] = zero_v8();

    float  aregf[16];
    uint4v aregb[2];
    float  bregf[16];

    auto load_tile = [&](int k0) {
        if constexpr (A_IS_BF16) {
            const unsigned short* Ab =
                (const unsigned short*)Aptr + (size_t)(m0 + ar) * K + k0 + acoff;
            aregb[0] = *(const uint4v*)Ab;
            aregb[1] = *(const uint4v*)(Ab + 8);
            if (k0 + BK < K) __builtin_prefetch(Ab + BK, 0, 1);
        } else {
            const float* Af = (const float*)Aptr + (size_t)(m0 + ar) * K + k0 + acoff;
            const float4* af = (const float4*)Af;
            float4 f0 = af[0], f1 = af[1], f2 = af[2], f3 = af[3];
            aregf[0]=f0.x; aregf[1]=f0.y; aregf[2]=f0.z; aregf[3]=f0.w;
            aregf[4]=f1.x; aregf[5]=f1.y; aregf[6]=f1.z; aregf[7]=f1.w;
            aregf[8]=f2.x; aregf[9]=f2.y; aregf[10]=f2.z; aregf[11]=f2.w;
            aregf[12]=f3.x; aregf[13]=f3.y; aregf[14]=f3.z; aregf[15]=f3.w;
            if (k0 + BK < K) __builtin_prefetch(Af + BK, 0, 1);
        }
        const float* Bf = B + (size_t)(k0 + bk) * N + n0 + bnoff;
        const float4* bf4 = (const float4*)Bf;
        float4 g0 = bf4[0], g1 = bf4[1], g2 = bf4[2], g3 = bf4[3];
        bregf[0]=g0.x; bregf[1]=g0.y; bregf[2]=g0.z; bregf[3]=g0.w;
        bregf[4]=g1.x; bregf[5]=g1.y; bregf[6]=g1.z; bregf[7]=g1.w;
        bregf[8]=g2.x; bregf[9]=g2.y; bregf[10]=g2.z; bregf[11]=g2.w;
        bregf[12]=g3.x; bregf[13]=g3.y; bregf[14]=g3.z; bregf[15]=g3.w;
        if (k0 + BK < K) __builtin_prefetch(Bf + (size_t)BK * N, 0, 1);
    };

    auto store_tile = [&](int buf) {
        if constexpr (A_IS_BF16) {
            *(uint4v*)&As[buf][ar][acoff]     = aregb[0];
            *(uint4v*)&As[buf][ar][acoff + 8] = aregb[1];
        } else {
            union { __bf16 h[16]; uint4v u[2]; } pk;
            #pragma unroll
            for (int e = 0; e < 16; ++e) pk.h[e] = (__bf16)aregf[e];
            *(uint4v*)&As[buf][ar][acoff]     = pk.u[0];
            *(uint4v*)&As[buf][ar][acoff + 8] = pk.u[1];
        }
        #pragma unroll
        for (int e = 0; e < 16; ++e) Bs[buf][bnoff + e][bk] = to_bf16(bregf[e]);
    };

    auto compute_tile = [&](int buf) {
        FragAB afr[2], bfr[4];
        #pragma unroll
        for (int i = 0; i < 2; ++i) {
            const unsigned short* p = &As[buf][wm + i * 16 + l16][half * 8];
            afr[i].u[0] = *(const uint4v*)p;
            afr[i].u[1] = *(const uint4v*)(p + 16);
        }
        #pragma unroll
        for (int j = 0; j < 4; ++j) {
            const unsigned short* p = &Bs[buf][wn + j * 16 + l16][half * 8];
            bfr[j].u[0] = *(const uint4v*)p;
            bfr[j].u[1] = *(const uint4v*)(p + 16);
        }
        #pragma unroll
        for (int i = 0; i < 2; ++i)
            #pragma unroll
            for (int j = 0; j < 4; ++j)
                acc[i][j] = wmma_bf16(afr[i], bfr[j], acc[i][j]);
    };

    const int nk = K / BK;
    load_tile(0);
    store_tile(0);
    __syncthreads();
    for (int kt = 0; kt < nk; ++kt) {
        const int cur = kt & 1;
        const bool more = (kt + 1) < nk;
        if (more) load_tile((kt + 1) * BK);     // global loads in flight...
        compute_tile(cur);                      // ...while WMMAs run
        if (more) store_tile(cur ^ 1);          // wait + convert + LDS store
        __syncthreads();
    }

    // ---- epilogue: C layout VGPR e -> rows e / e+8 per half ----
    #pragma unroll
    for (int i = 0; i < 2; ++i)
        #pragma unroll
        for (int j = 0; j < 4; ++j)
            #pragma unroll
            for (int e = 0; e < 8; ++e) {
                const int r = m0 + wm + i * 16 + e + (half ? 8 : 0);
                const int c = n0 + wn + j * 16 + l16;
                const float val = acc[i][j][e];
                if constexpr (OUT_BF16) {
                    ((unsigned short*)Cptr)[(size_t)r * N + c] = to_bf16(val);
                } else {
                    int orow = r;
                    if constexpr (REMAP_BL) orow = (r & (L_SEQ - 1)) * BATCH + (r >> 11);
                    ((float*)Cptr)[(size_t)orow * N + c] = val;
                }
            }
}

// ---------------------------------------------------------------------------
// RoPE (q scaled by 1/sqrt(D)) + scatter q->[B,NH,L,D], k->[B,NKV,L,D]
// one thread per (row, head, freq-index)
// ---------------------------------------------------------------------------
__global__ __launch_bounds__(256)
void rope_scatter_kernel(const unsigned short* __restrict__ qkv,
                         const int* __restrict__ pos_ids,
                         unsigned short* __restrict__ Qr,
                         unsigned short* __restrict__ Kr) {
    const int idx  = blockIdx.x * blockDim.x + threadIdx.x;
    const int i    = idx & 63;          // frequency index 0..63
    const int t    = idx >> 6;
    const int head = t % (NH + NKV);    // 0..31 = q heads, 32..39 = k heads
    const int r    = t / (NH + NKV);    // row in [0, ROWS)
    const int l = r >> 1, b = r & 1;    // r = l*B + b

    const bool isq = head < NH;
    const int  col0 = isq ? (head * HD + i) : (NH * HD + (head - NH) * HD + i);
    const float x0 = bf16_to_f32(qkv[(size_t)r * QKVN + col0]);
    const float x1 = bf16_to_f32(qkv[(size_t)r * QKVN + col0 + 64]);

    const float pos  = (float)pos_ids[b * L_SEQ + l];
    const float invf = __expf(-(float)i * 0.14391156955f);   // ln(10000)/64
    float s, c;
    __sincosf(pos * invf, &s, &c);
    float y0 = x0 * c - x1 * s;
    float y1 = x1 * c + x0 * s;
    if (isq) {
        y0 *= ATTN_SCALE; y1 *= ATTN_SCALE;
        const size_t base = (((size_t)b * NH + head) * L_SEQ + l) * HD;
        Qr[base + i] = to_bf16(y0); Qr[base + i + 64] = to_bf16(y1);
    } else {
        const size_t base = (((size_t)b * NKV + (head - NH)) * L_SEQ + l) * HD;
        Kr[base + i] = to_bf16(y0); Kr[base + i + 64] = to_bf16(y1);
    }
}

// V scatter: qkv cols [5120, 6144) -> [B, NKV, L, D]
__global__ __launch_bounds__(256)
void copy_v_kernel(const unsigned short* __restrict__ qkv,
                   unsigned short* __restrict__ Vr) {
    const int idx = blockIdx.x * blockDim.x + threadIdx.x;
    const int c = idx & 1023;
    const int r = idx >> 10;
    const int l = r >> 1, b = r & 1;
    const int kh = c >> 7, d = c & 127;
    Vr[(((size_t)b * NKV + kh) * L_SEQ + l) * HD + d] =
        qkv[(size_t)r * QKVN + (NH * HD + NKV * HD) + c];
}

// ---------------------------------------------------------------------------
// Causal flash attention, one block per (b,h, 128-row q-tile).
// 8 waves; wave w owns q rows [qb*128 + w*16, +16). KV tiles of 32 keys.
// QK^T and P*V both via v_wmma_f32_16x16x32_bf16. Online softmax in f32.
// ---------------------------------------------------------------------------
__global__ __launch_bounds__(256)
void flash_attn_kernel(const unsigned short* __restrict__ Qr,
                       const unsigned short* __restrict__ Kr,
                       const unsigned short* __restrict__ Vr,
                       unsigned short* __restrict__ O16) {
    __shared__ unsigned short Ks[32][HD];       // keys row-major       8 KB
    __shared__ unsigned short Vt[HD][32];       // V transposed [d][k]  8 KB
    __shared__ unsigned short Ps[8][16][32];    // per-wave P tile      8 KB

    const int tid  = threadIdx.x;
    const int wave = tid >> 5;
    const int lane = tid & 31;
    const int half = lane >> 4;
    const int l16  = lane & 15;

    const int qb  = blockIdx.x;              // q tile (128 rows)
    const int bh  = blockIdx.y;
    const int b   = bh >> 5, h = bh & 31;
    const int kvh = h >> 2;                  // GQA: 4 q heads per kv head

    const size_t qbase = ((size_t)b * NH + h)    * (size_t)L_SEQ * HD;
    const size_t kbase = ((size_t)b * NKV + kvh) * (size_t)L_SEQ * HD;
    const int qrow0 = qb * 128 + wave * 16;

    // Q fragments resident in registers: 4 x (16x32) covering D=128
    FragAB qf[4];
    {
        const unsigned short* qp = Qr + qbase + (size_t)(qrow0 + l16) * HD;
        #pragma unroll
        for (int f = 0; f < 4; ++f) {
            const unsigned short* p = qp + f * 32 + half * 8;
            qf[f].u[0] = *(const uint4v*)p;
            qf[f].u[1] = *(const uint4v*)(p + 16);
        }
    }

    float m_run[8], l_run[8];
    v8f oacc[8];
    #pragma unroll
    for (int e = 0; e < 8; ++e) { m_run[e] = NEG_BIG; l_run[e] = 0.0f; oacc[e] = zero_v8(); }

    const int nkt = qb * 4 + 4;              // causal bound for this block
    for (int kt = 0; kt < nkt; ++kt) {
        __syncthreads();
        // stage K tile (row-major) and V tile (transposed)
        {
            const int kr = tid >> 3, doff = (tid & 7) * 16;
            const unsigned short* ksrc = Kr + kbase + (size_t)(kt * 32 + kr) * HD + doff;
            *(uint4v*)&Ks[kr][doff]     = *(const uint4v*)ksrc;
            *(uint4v*)&Ks[kr][doff + 8] = *(const uint4v*)(ksrc + 8);
            const unsigned short* vsrc = Vr + kbase + (size_t)(kt * 32 + kr) * HD + doff;
            #pragma unroll
            for (int e = 0; e < 16; ++e) Vt[doff + e][kr] = vsrc[e];
        }
        __syncthreads();

        // S = Q K^T  (16 q rows x 32 keys)  -> 8 WMMA over D
        v8f s0 = zero_v8(), s1 = zero_v8();
        #pragma unroll
        for (int f = 0; f < 4; ++f) {
            FragAB kb0, kb1;
            const unsigned short* p0 = &Ks[l16][f * 32 + half * 8];
            kb0.u[0] = *(const uint4v*)p0; kb0.u[1] = *(const uint4v*)(p0 + 16);
            const unsigned short* p1 = &Ks[16 + l16][f * 32 + half * 8];
            kb1.u[0] = *(const uint4v*)p1; kb1.u[1] = *(const uint4v*)(p1 + 16);
            s0 = wmma_bf16(qf[f], kb0, s0);
            s1 = wmma_bf16(qf[f], kb1, s1);
        }

        // causal mask + online softmax update
        float alpha[8];
        #pragma unroll
        for (int e = 0; e < 8; ++e) {
            const int rq = qrow0 + e + half * 8;
            const int c0 = kt * 32 + l16;
            float v0 = (c0 <= rq)      ? s0[e] : NEG_BIG;
            float v1 = (c0 + 16 <= rq) ? s1[e] : NEG_BIG;
            float m  = red_max16(fmaxf(v0, v1));
            const float mnew = fmaxf(m_run[e], m);
            alpha[e] = __expf(m_run[e] - mnew);
            const float p0 = __expf(v0 - mnew);
            const float p1 = __expf(v1 - mnew);
            l_run[e] = l_run[e] * alpha[e] + red_sum16(p0 + p1);
            m_run[e] = mnew;
            Ps[wave][e + half * 8][l16]      = to_bf16(p0);
            Ps[wave][e + half * 8][l16 + 16] = to_bf16(p1);
        }
        #pragma unroll
        for (int j = 0; j < 8; ++j)
            #pragma unroll
            for (int e = 0; e < 8; ++e) oacc[j][e] *= alpha[e];

        // O += P V  (A-frag from per-wave LDS re-layout; B-frag from Vt)
        FragAB pf;
        const unsigned short* pp = &Ps[wave][l16][half * 8];
        pf.u[0] = *(const uint4v*)pp;
        pf.u[1] = *(const uint4v*)(pp + 16);
        #pragma unroll
        for (int j = 0; j < 8; ++j) {
            FragAB vf;
            const unsigned short* vp = &Vt[j * 16 + l16][half * 8];
            vf.u[0] = *(const uint4v*)vp;
            vf.u[1] = *(const uint4v*)(vp + 16);
            oacc[j] = wmma_bf16(pf, vf, oacc[j]);
        }
    }

    // finalize: O16[b*L + rq][h*128 + d]
    float inv_l[8];
    #pragma unroll
    for (int e = 0; e < 8; ++e) inv_l[e] = 1.0f / l_run[e];
    #pragma unroll
    for (int j = 0; j < 8; ++j)
        #pragma unroll
        for (int e = 0; e < 8; ++e) {
            const int rq = qrow0 + e + half * 8;
            const int d  = j * 16 + l16;
            O16[((size_t)b * L_SEQ + rq) * (size_t)(NH * HD) + h * HD + d] =
                to_bf16(oacc[j][e] * inv_l[e]);
        }
}

// ---------------------------------------------------------------------------
// Host launch
// ---------------------------------------------------------------------------
extern "C" void kernel_launch(void* const* d_in, const int* in_sizes, int n_in,
                              void* d_out, int out_size, void* d_ws, size_t ws_size,
                              hipStream_t stream) {
    (void)in_sizes; (void)n_in; (void)out_size; (void)ws_size;
    const float* hidden = (const float*)d_in[0];   // [L,B,HID] fp32
    const int*   pos    = (const int*)d_in[1];     // [B,L] int32
    const float* Wqkv   = (const float*)d_in[2];   // [HID, 6144] fp32
    const float* Wo     = (const float*)d_in[3];   // [4096, HID] fp32
    float* out = (float*)d_out;                    // [L,B,HID] fp32

    char* ws = (char*)d_ws;
    unsigned short* QKV16 = (unsigned short*)(ws);                 // 48 MB
    unsigned short* Qrb   = (unsigned short*)(ws + 50331648);      // 32 MB
    unsigned short* Krb   = (unsigned short*)(ws + 83886080);      //  8 MB
    unsigned short* Vrb   = (unsigned short*)(ws + 92274688);      //  8 MB
    unsigned short* O16   = (unsigned short*)(ws + 100663296);     // 32 MB

    const dim3 blk(256);
    // 1) QKV projection: [4096,4096] x [4096,6144] -> bf16 [4096,6144]
    gemm_bf16_wmma<false, true, false>
        <<<dim3(QKVN / 128, ROWS / 128), blk, 0, stream>>>(hidden, Wqkv, QKV16,
                                                           ROWS, QKVN, HIDDIM);
    // 2) RoPE + scatter q/k, copy v
    rope_scatter_kernel<<<(ROWS * (NH + NKV) * 64) / 256, blk, 0, stream>>>(QKV16, pos, Qrb, Krb);
    copy_v_kernel<<<(ROWS * NKV * HD) / 256, blk, 0, stream>>>(QKV16, Vrb);
    // 3) causal flash attention -> bf16 [B*L, H*D]
    flash_attn_kernel<<<dim3(L_SEQ / 128, BATCH * NH), blk, 0, stream>>>(Qrb, Krb, Vrb, O16);
    // 4) output projection with (b*L+l) -> (l*B+b) row remap, fp32 out
    gemm_bf16_wmma<true, false, true>
        <<<dim3(HIDDIM / 128, ROWS / 128), blk, 0, stream>>>(O16, Wo, out,
                                                             ROWS, HIDDIM, NH * HD);
}
